// PDGNNDecoder_27290222198799
// MI455X (gfx1250) — matrix operand order repaired and checked by reading, not compiled, not run on hardware
//
#include <hip/hip_runtime.h>
#include <math.h>

// ---------------------------------------------------------------------------
// PDGNN decoder for MI455X (gfx1250, wave32).
// Roofline: edge aggregation (3 layers x 1.7M edges x 512B gather + 512B
// atomic-add) dominates; m/agg (51MB each) are L2-resident (192MB L2), so the
// gathers run at L2 bandwidth and fadd atomics use the L2 atomic units.
// Dense GEMMs (~18 GFLOP) run on v_wmma_f32_16x16x32_bf16 with LDS-staged,
// fragment-swizzled tiles (packed b32 staging stores). All elementwise work
// (self-loop, bias, LayerNorm, ReLU, sigmoid-gate / residual) is fused into a
// single per-node wave32 kernel to avoid extra 51MB passes.
// ---------------------------------------------------------------------------

typedef __attribute__((ext_vector_type(16))) __bf16 v16bf;
typedef __attribute__((ext_vector_type(8)))  float  v8f;

__device__ __forceinline__ void atomAddF(float* p, float v) {
  // relaxed agent-scope fadd -> native global_atomic_add_f32 on gfx1250
  __hip_atomic_fetch_add(p, v, __ATOMIC_RELAXED, __HIP_MEMORY_SCOPE_AGENT);
}

__device__ __forceinline__ unsigned pack2_bf16(float a, float b) {
  unsigned short u0 = __builtin_bit_cast(unsigned short, (__bf16)a);
  unsigned short u1 = __builtin_bit_cast(unsigned short, (__bf16)b);
  return (unsigned)u0 | ((unsigned)u1 << 16);
}

// ------------------------- graph preprocessing -----------------------------

__global__ __launch_bounds__(256)
void deg_kernel(const long long* __restrict__ dst, int E, unsigned* __restrict__ deg) {
  int e = blockIdx.x * 256 + threadIdx.x;
  if (e < E) atomicAdd(&deg[(int)dst[e]], 1u);
}

__global__ __launch_bounds__(256)
void dis_kernel(const unsigned* __restrict__ deg, float* __restrict__ dis, int N) {
  int i = blockIdx.x * 256 + threadIdx.x;
  if (i < N) dis[i] = rsqrtf((float)(deg[i] + 1u));   // +1 = self loop, deg>=1
}

__global__ __launch_bounds__(256)
void norm_kernel(const long long* __restrict__ src, const long long* __restrict__ dst,
                 const float* __restrict__ dis, float* __restrict__ norm, int E) {
  int e = blockIdx.x * 256 + threadIdx.x;
  if (e < E) norm[e] = dis[(int)src[e]] * dis[(int)dst[e]];
}

// concat [x | pk_emb | pk_pred | zero-pad] -> fp32 [N][224]
__global__ __launch_bounds__(256)
void concat_kernel(const float* __restrict__ x, const float* __restrict__ emb,
                   const float* __restrict__ pred, float* __restrict__ h0, int total) {
  int t = blockIdx.x * 256 + threadIdx.x;
  if (t >= total) return;
  int n = t / 224, c = t - n * 224;
  float v;
  if      (c < 128) v = x[n * 128 + c];
  else if (c < 192) v = emb[n * 64 + (c - 128)];
  else if (c == 192) v = pred[n];
  else               v = 0.f;
  h0[t] = v;
}

// ------------------------- WMMA bf16 GEMM ----------------------------------
// C[N x 128] = A[N x KPAD] * W[krows x 128]   (KPAD = KCH*32, zero-padded)
// Block: 256 thr = 8 waves; each wave -> 16 rows x 128 cols.
// LDS holds A and W pre-swizzled into CDNA5 WMMA fragment layouts so every
// fragment read is one contiguous 32B ds load. Staging packs two consecutive
// k (same lane, adjacent j in both fragment layouts) into one b32 store.

template <int KCH>
__global__ __launch_bounds__(256)
void gemm_bf16_kernel(const float* __restrict__ A, int lda,
                      const float* __restrict__ W, int krows,
                      float* __restrict__ C, int N) {
  constexpr int KPAD = KCH * 32;
  extern __shared__ __align__(32) char smem_raw[];
  __bf16* aLds = (__bf16*)smem_raw;          // 8 waves * KCH * 512 elems
  __bf16* wLds = aLds + 8 * KCH * 512;       // KCH * 8 nblk * 512 elems
  unsigned* aLds32 = (unsigned*)aLds;
  unsigned* wLds32 = (unsigned*)wLds;

  const int tid  = threadIdx.x;
  const int wave = tid >> 5;
  const int lane = tid & 31;
  const int rowBase = blockIdx.x * 128;

  // Stage W (fp32 -> bf16), B-fragment layout (32x16 per frag):
  //   lane = (k%32)/16*16 + col%16 ; j = k%16. Pair (k, k+1) -> one b32.
  for (int idx = tid; idx < KPAD * 64; idx += 256) {
    int k = (idx >> 7) << 1;          // even k
    int n = idx & 127;
    float v0 = (k     < krows) ? W[k * 128 + n]       : 0.f;
    float v1 = (k + 1 < krows) ? W[(k + 1) * 128 + n] : 0.f;
    int kc = k >> 5, kk = k & 31;
    int lw = (kk >> 4) * 16 + (n & 15);
    int j  = kk & 15;                 // even
    wLds32[(((kc * 8 + (n >> 4)) * 32 + lw) * 16 + j) >> 1] = pack2_bf16(v0, v1);
  }
  // Stage A tile (128 rows), A-fragment layout (16x32 per frag):
  //   lane = (k%32)/16*16 + row%16 ; j = k%16
  //   Pair (k, k+1), k even -> same lane, j even -> one b32 store.
  for (int idx = tid; idx < 64 * KPAD; idx += 256) {
    int r = idx / (KPAD / 2);
    int k = (idx - r * (KPAD / 2)) << 1;   // even k
    int row = rowBase + r;
    float2 v = make_float2(0.f, 0.f);
    if (row < N) v = *(const float2*)(A + (long long)row * lda + k);
    int kc = k >> 5, kk = k & 31;
    int la = ((kk >> 4) * 16) + (r & 15);
    int j  = kk & 15;   // even
    aLds32[((((r >> 4) * KCH + kc) * 32 + la) * 16 + j) >> 1] = pack2_bf16(v.x, v.y);
  }
  __syncthreads();

  v8f acc[8] = {};
#pragma unroll
  for (int kc = 0; kc < KCH; ++kc) {
    v16bf a = *(const v16bf*)(aLds + ((wave * KCH + kc) * 32 + lane) * 16);
#pragma unroll
    for (int nb = 0; nb < 8; ++nb) {
      v16bf b = *(const v16bf*)(wLds + ((kc * 8 + nb) * 32 + lane) * 16);
      acc[nb] = __builtin_amdgcn_wmma_f32_16x16x32_bf16(
          false, a, false, b, (short)0, acc[nb], false, false);
    }
  }

  // C/D layout: col = lane%16 ; row = wave*16 + (lane/16)*8 + vgpr
  const int colLo = lane & 15;
  const int rOff  = (lane >> 4) * 8;
#pragma unroll
  for (int nb = 0; nb < 8; ++nb) {
#pragma unroll
    for (int v = 0; v < 8; ++v) {
      int row = rowBase + wave * 16 + rOff + v;
      if (row < N) C[(long long)row * 128 + nb * 16 + colLo] = acc[nb][v];
    }
  }
}

// ------------------------- edge aggregation --------------------------------

// agg[dst] += m[src] * norm[e]   -- one wave per edge, float4 per lane
__global__ __launch_bounds__(256)
void scatter_kernel(const float* __restrict__ m, const long long* __restrict__ src,
                    const long long* __restrict__ dst, const float* __restrict__ norm,
                    float* __restrict__ agg, int E) {
  int t = blockIdx.x * 256 + threadIdx.x;
  int e = t >> 5, lane = t & 31;
  if (e >= E) return;
  int s = (int)src[e], d = (int)dst[e];
  float w = norm[e];
  float4 mv = *(const float4*)(m + (long long)s * 128 + lane * 4);
  float* ap = agg + (long long)d * 128 + lane * 4;
  atomAddF(ap + 0, mv.x * w);
  atomAddF(ap + 1, mv.y * w);
  atomAddF(ap + 2, mv.z * w);
  atomAddF(ap + 3, mv.w * w);
}

// ------------------------- fused node epilogue -----------------------------
// v = agg + m*dis^2 + bias        (self-loop + bias folded in)
// hn = relu(LN(v; g, be))
// mode 0: hout = hn * sigmoid(other + bg)   (other = gate-GEMM output)
// mode 1: hout = other + hn                 (residual; other == hout ok)
// one node per wave (32 lanes x float4 = 128), wave32 shuffle reductions
__global__ __launch_bounds__(256)
void post_kernel(const float* __restrict__ agg, const float* __restrict__ m,
                 const float* __restrict__ dis, const float* __restrict__ bias,
                 const float* __restrict__ g, const float* __restrict__ be,
                 const float* __restrict__ other, const float* __restrict__ bg,
                 float* __restrict__ hout, int N, int mode) {
  int wave = threadIdx.x >> 5, lane = threadIdx.x & 31;
  int n = blockIdx.x * 8 + wave;
  if (n >= N) return;
  long long base = (long long)n * 128 + lane * 4;
  float di = dis[n];
  float sl = di * di;
  float4 v  = *(const float4*)(agg + base);
  float4 mv = *(const float4*)(m + base);
  float4 bb = *(const float4*)(bias + lane * 4);
  float4 hv = make_float4(v.x + mv.x * sl + bb.x, v.y + mv.y * sl + bb.y,
                          v.z + mv.z * sl + bb.z, v.w + mv.w * sl + bb.w);
  float s  = hv.x + hv.y + hv.z + hv.w;
  float s2 = hv.x * hv.x + hv.y * hv.y + hv.z * hv.z + hv.w * hv.w;
  for (int o = 16; o; o >>= 1) { s += __shfl_xor(s, o, 32); s2 += __shfl_xor(s2, o, 32); }
  float mu  = s * (1.f / 128.f);
  float var = s2 * (1.f / 128.f) - mu * mu;
  float rs  = rsqrtf(var + 1e-5f);
  float4 gg  = *(const float4*)(g  + lane * 4);
  float4 bbe = *(const float4*)(be + lane * 4);
  float4 o4  = *(const float4*)(other + base);
  float4 r;
  r.x = fmaxf((hv.x - mu) * rs * gg.x + bbe.x, 0.f);
  r.y = fmaxf((hv.y - mu) * rs * gg.y + bbe.y, 0.f);
  r.z = fmaxf((hv.z - mu) * rs * gg.z + bbe.z, 0.f);
  r.w = fmaxf((hv.w - mu) * rs * gg.w + bbe.w, 0.f);
  if (mode == 0) {
    float4 bgv = *(const float4*)(bg + lane * 4);
    r.x *= 1.f / (1.f + expf(-(o4.x + bgv.x)));
    r.y *= 1.f / (1.f + expf(-(o4.y + bgv.y)));
    r.z *= 1.f / (1.f + expf(-(o4.z + bgv.z)));
    r.w *= 1.f / (1.f + expf(-(o4.w + bgv.w)));
  } else {
    r.x += o4.x; r.y += o4.y; r.z += o4.z; r.w += o4.w;
  }
  *(float4*)(hout + base) = r;
}

// head: out = relu(h @ Wp1 + bp1) @ Wp2 + bp2 ; one node per wave
__global__ __launch_bounds__(256)
void head_kernel(const float* __restrict__ h, const float* __restrict__ Wp1,
                 const float* __restrict__ bp1, const float* __restrict__ Wp2,
                 const float* __restrict__ bp2, float* __restrict__ out, int N) {
  __shared__ float w1[128 * 64];
  __shared__ float w2[64];
  __shared__ float rows[8][128];
  int tid = threadIdx.x;
  for (int i = tid; i < 128 * 64; i += 256) w1[i] = Wp1[i];
  if (tid < 64) w2[tid] = Wp2[tid];
  int wave = tid >> 5, lane = tid & 31;
  int n = blockIdx.x * 8 + wave;
  bool valid = n < N;
  if (valid) {
    float4 hv = *(const float4*)(h + (long long)n * 128 + lane * 4);
    *(float4*)(&rows[wave][lane * 4]) = hv;
  }
  __syncthreads();
  if (!valid) return;
  float a0 = bp1[lane], a1 = bp1[lane + 32];
#pragma unroll 4
  for (int k = 0; k < 128; ++k) {
    float hv = rows[wave][k];
    a0 = fmaf(hv, w1[k * 64 + lane], a0);
    a1 = fmaf(hv, w1[k * 64 + lane + 32], a1);
  }
  a0 = fmaxf(a0, 0.f);
  a1 = fmaxf(a1, 0.f);
  float p = a0 * w2[lane] + a1 * w2[lane + 32];
  for (int o = 16; o; o >>= 1) p += __shfl_xor(p, o, 32);
  if (lane == 0) out[n] = p + bp2[0];
}

// ------------------------------- launch ------------------------------------

static inline unsigned cdiv(long long a, long long b) { return (unsigned)((a + b - 1) / b); }

extern "C" void kernel_launch(void* const* d_in, const int* in_sizes, int n_in,
                              void* d_out, int out_size, void* d_ws, size_t ws_size,
                              hipStream_t stream) {
  const float* x    = (const float*)d_in[0];
  const float* emb  = (const float*)d_in[1];
  const float* pred = (const float*)d_in[2];
  const long long* ei = (const long long*)d_in[3];
  const float* W0 = (const float*)d_in[4];  const float* b0 = (const float*)d_in[5];
  const float* W1 = (const float*)d_in[6];  const float* b1 = (const float*)d_in[7];
  const float* W2 = (const float*)d_in[8];  const float* b2 = (const float*)d_in[9];
  const float* g0 = (const float*)d_in[10]; const float* be0 = (const float*)d_in[11];
  const float* g1 = (const float*)d_in[12]; const float* be1 = (const float*)d_in[13];
  const float* g2 = (const float*)d_in[14]; const float* be2 = (const float*)d_in[15];
  const float* Wg = (const float*)d_in[16]; const float* bg = (const float*)d_in[17];
  const float* Wp1 = (const float*)d_in[18]; const float* bp1 = (const float*)d_in[19];
  const float* Wp2 = (const float*)d_in[20]; const float* bp2 = (const float*)d_in[21];

  const int N = in_sizes[0] / 128;
  const int E = in_sizes[3] / 2;
  const long long* srcp = ei;
  const long long* dstp = ei + E;

  // carve workspace
  char* ws = (char*)d_ws;
  auto carve = [&](size_t bytes) {
    char* p = ws; ws += (bytes + 255) & ~(size_t)255; return p;
  };
  unsigned* deg = (unsigned*)carve((size_t)N * 4);
  float* dis  = (float*)carve((size_t)N * 4);
  float* norm = (float*)carve((size_t)E * 4);
  float* h0   = (float*)carve((size_t)N * 224 * 4);
  float* gate = (float*)carve((size_t)N * 128 * 4);  // raw gate-GEMM output
  float* m    = (float*)carve((size_t)N * 128 * 4);
  float* agg  = (float*)carve((size_t)N * 128 * 4);
  float* h    = (float*)carve((size_t)N * 128 * 4);

  const unsigned gGemm = cdiv(N, 128);
  const unsigned gNode = cdiv(N, 8);
  const unsigned gEdgeW = cdiv((long long)E * 32, 256);
  const size_t lds7 = 7 * 16384;  // KCH=7: A 8*7*512 + W 7*8*512 bf16
  const size_t lds4 = 4 * 16384;  // KCH=4

  // graph preprocessing
  hipMemsetAsync(deg, 0, (size_t)N * 4, stream);
  deg_kernel<<<cdiv(E, 256), 256, 0, stream>>>(dstp, E, deg);
  dis_kernel<<<cdiv(N, 256), 256, 0, stream>>>(deg, dis, N);
  norm_kernel<<<cdiv(E, 256), 256, 0, stream>>>(srcp, dstp, dis, norm, E);
  concat_kernel<<<cdiv((long long)N * 224, 256), 256, 0, stream>>>(x, emb, pred, h0,
                                                                   N * 224);
  // gate linear (sigmoid fused into layer-0 post)
  gemm_bf16_kernel<7><<<gGemm, 256, lds7, stream>>>(h0, 224, Wg, 193, gate, N);

  // layer 0 (gated, no residual)
  gemm_bf16_kernel<7><<<gGemm, 256, lds7, stream>>>(h0, 224, W0, 193, m, N);
  hipMemsetAsync(agg, 0, (size_t)N * 128 * 4, stream);
  scatter_kernel<<<gEdgeW, 256, 0, stream>>>(m, srcp, dstp, norm, agg, E);
  post_kernel<<<gNode, 256, 0, stream>>>(agg, m, dis, b0, g0, be0, gate, bg, h, N, 0);

  // layer 1 (residual)
  gemm_bf16_kernel<4><<<gGemm, 256, lds4, stream>>>(h, 128, W1, 128, m, N);
  hipMemsetAsync(agg, 0, (size_t)N * 128 * 4, stream);
  scatter_kernel<<<gEdgeW, 256, 0, stream>>>(m, srcp, dstp, norm, agg, E);
  post_kernel<<<gNode, 256, 0, stream>>>(agg, m, dis, b1, g1, be1, h, bg, h, N, 1);

  // layer 2 (residual)
  gemm_bf16_kernel<4><<<gGemm, 256, lds4, stream>>>(h, 128, W2, 128, m, N);
  hipMemsetAsync(agg, 0, (size_t)N * 128 * 4, stream);
  scatter_kernel<<<gEdgeW, 256, 0, stream>>>(m, srcp, dstp, norm, agg, E);
  post_kernel<<<gNode, 256, 0, stream>>>(agg, m, dis, b2, g2, be2, h, bg, h, N, 1);

  // head
  head_kernel<<<gNode, 256, 0, stream>>>(h, Wp1, bp1, Wp2, bp2, (float*)d_out, N);
}
